// TrajectoryAttentionBlock_60413009985972
// MI455X (gfx1250) — compile-verified, matrix operand order checked by reading, not run on hardware
//
#include <hip/hip_runtime.h>

typedef unsigned short u16;
typedef unsigned int   u32;

typedef __attribute__((ext_vector_type(16))) __bf16 bf16x16;
typedef __attribute__((ext_vector_type(8)))  float  f32x8;
typedef int v4i __attribute__((__vector_size__(16)));   // matches builtin's V4i

#define __AS1 __attribute__((address_space(1)))
#define __AS3 __attribute__((address_space(3)))

#if __has_builtin(__builtin_amdgcn_global_load_async_to_lds_b128) && \
    __has_builtin(__builtin_amdgcn_s_wait_asynccnt)
#define USE_ASYNC_LDS 1
#else
#define USE_ASYNC_LDS 0
#endif

union AFrag {
  bf16x16 v;
  uint4   q[2];
  u16     us[16];
};

__device__ __forceinline__ u16 f2bf(float f) {
  u32 u = __float_as_uint(f);
  u += 0x7fffu + ((u >> 16) & 1u);       // round-to-nearest-even
  return (u16)(u >> 16);
}

__device__ __forceinline__ f32x8 wmma_bf16(bf16x16 a, bf16x16 b, f32x8 c) {
  // D = A(16x32 bf16) * B(32x16 bf16) + C(16x16 f32)
  return __builtin_amdgcn_wmma_f32_16x16x32_bf16(false, a, false, b, (short)0, c, false, false);
}

__device__ __forceinline__ float rmax16(float v) {
  v = fmaxf(v, __shfl_xor(v, 1));
  v = fmaxf(v, __shfl_xor(v, 2));
  v = fmaxf(v, __shfl_xor(v, 4));
  v = fmaxf(v, __shfl_xor(v, 8));
  return v;
}
__device__ __forceinline__ float rsum16(float v) {
  v += __shfl_xor(v, 1);
  v += __shfl_xor(v, 2);
  v += __shfl_xor(v, 4);
  v += __shfl_xor(v, 8);
  return v;
}

// ---------------------------------------------------------------- fp32 -> bf16
__global__ __launch_bounds__(256) void k_cvt(const float* __restrict__ in,
                                             u16* __restrict__ outp, int n) {
  int i = blockIdx.x * 256 + threadIdx.x;
  if (i < n) outp[i] = f2bf(in[i]);
}

// ---------------------------------------------------------------- LayerNorm (D=768)
// mode 0: src = src0 + r*768                    (direct, r over M rows)
// mode 1: r=(b,i) over (8,3136): src = src0[b, 1+i]  (x[:,1:])
// mode 2: r=(b,t,j) over (8,16,197): j==0 -> src0[b,0] (CLS), j>0 -> src1[b, (j-1)*16+t]
__global__ __launch_bounds__(256)
void k_ln(const float* __restrict__ src0, const float* __restrict__ src1,
          const float* __restrict__ g, const float* __restrict__ bta,
          u16* __restrict__ outp, int mode) {
  int r = blockIdx.x;
  const float* src;
  if (mode == 0) {
    src = src0 + (size_t)r * 768;
  } else if (mode == 1) {
    int b = r / 3136, i = r - b * 3136;
    src = src0 + ((size_t)b * 3137 + 1 + i) * 768;
  } else {
    int seq = r / 197, j = r - seq * 197;
    int b = seq >> 4, t = seq & 15;
    if (j == 0) src = src0 + (size_t)b * 3137 * 768;
    else        src = src1 + ((size_t)b * 3136 + (size_t)(j - 1) * 16 + t) * 768;
  }
  int tid = threadIdx.x;
  int wv = tid >> 5, ln = tid & 31;
  float v0 = src[tid], v1 = src[tid + 256], v2 = src[tid + 512];
  __shared__ float red[8];
  float s = v0 + v1 + v2;
  s += __shfl_xor(s, 1);  s += __shfl_xor(s, 2);  s += __shfl_xor(s, 4);
  s += __shfl_xor(s, 8);  s += __shfl_xor(s, 16);
  if (ln == 0) red[wv] = s;
  __syncthreads();
  float tot = 0.f;
#pragma unroll
  for (int i = 0; i < 8; i++) tot += red[i];
  float mean = tot * (1.0f / 768.0f);
  float d0 = v0 - mean, d1 = v1 - mean, d2 = v2 - mean;
  float s2 = d0 * d0 + d1 * d1 + d2 * d2;
  s2 += __shfl_xor(s2, 1); s2 += __shfl_xor(s2, 2); s2 += __shfl_xor(s2, 4);
  s2 += __shfl_xor(s2, 8); s2 += __shfl_xor(s2, 16);
  __syncthreads();
  if (ln == 0) red[wv] = s2;
  __syncthreads();
  float var = 0.f;
#pragma unroll
  for (int i = 0; i < 8; i++) var += red[i];
  var *= (1.0f / 768.0f);
  float rstd = rsqrtf(var + 1e-5f);
  u16* o = outp + (size_t)r * 768;
  o[tid]       = f2bf(d0 * rstd * g[tid]       + bta[tid]);
  o[tid + 256] = f2bf(d1 * rstd * g[tid + 256] + bta[tid + 256]);
  o[tid + 512] = f2bf(d2 * rstd * g[tid + 512] + bta[tid + 512]);
}

// ---------------------------------------------------------------- WMMA GEMM
// C(MxN) = A(MxK bf16 rowmajor) x B(KxN bf16 rowmajor) [+bias][+gelu][+res]
// 128x128 block, BK=32, 256 threads = 8 waves (2x4), wave tile 64x32.
// modes: 0 = bf16 out, 1 = bias+bf16, 2 = bias+gelu+bf16, 3 = bias+f32, 5 = bias+res+f32
// NOTE: A operand buffers are sized so the 128-row tail tile stays in-bounds;
// garbage rows only affect their own (unstored) C rows, so async staging needs
// no M-guard or zero-fill.
__global__ __launch_bounds__(256)
void k_gemm(const u16* __restrict__ A, const u16* __restrict__ Bw,
            const float* __restrict__ bias, const float* __restrict__ res,
            void* __restrict__ Cout, int M, int N, int K, int mode) {
  __shared__ u16 As[128 * 40];   // row-major [row][k], row stride 40 (80B)
  __shared__ u16 Bs[128 * 40];   // transposed [col][k], row stride 40
  const int bm = blockIdx.x * 128;
  const int bn = blockIdx.y * 128;
  const int tid = threadIdx.x;
  const int wave = tid >> 5, lane = tid & 31;
  const int wm = wave >> 2, wn = wave & 3;
  const int hf = lane >> 4, l16 = lane & 15;

  f32x8 acc[4][2];
  const f32x8 z8 = {0.f, 0.f, 0.f, 0.f, 0.f, 0.f, 0.f, 0.f};
#pragma unroll
  for (int mt = 0; mt < 4; mt++)
#pragma unroll
    for (int nt = 0; nt < 2; nt++) acc[mt][nt] = z8;

  for (int k0 = 0; k0 < K; k0 += 32) {
#if USE_ASYNC_LDS
    { // stage A via async global->LDS DMA (ASYNCcnt path)
      int row = tid >> 1, h = tid & 1;
      int grow = bm + row;
      const u16* gsrc = A + (size_t)grow * K + k0 + h * 16;
      u16* ldst = &As[row * 40 + h * 16];
      __builtin_amdgcn_global_load_async_to_lds_b128(
          (__AS1 v4i*)gsrc, (__AS3 v4i*)ldst, 0, 0);
      __builtin_amdgcn_global_load_async_to_lds_b128(
          (__AS1 v4i*)(gsrc + 8), (__AS3 v4i*)(ldst + 8), 0, 0);
    }
#else
    { // stage A through VGPRs
      int row = tid >> 1, h = tid & 1;
      int grow = bm + row;
      uint4 d0, d1;
      d0.x = d0.y = d0.z = d0.w = 0u;
      d1 = d0;
      if (grow < M) {
        const uint4* src = (const uint4*)(A + (size_t)grow * K + k0 + h * 16);
        d0 = src[0];
        d1 = src[1];
        if (k0 + 32 < K) __builtin_prefetch(A + (size_t)grow * K + k0 + 32 + h * 16, 0, 1);
      }
      *(uint4*)&As[row * 40 + h * 16]     = d0;
      *(uint4*)&As[row * 40 + h * 16 + 8] = d1;
    }
#endif
    { // stage B transposed: thread -> (k = tid>>3, colgroup = (tid&7)*16)
      int kk = tid >> 3;
      int cg = (tid & 7) << 4;
      const u16* src = Bw + (size_t)(k0 + kk) * N + bn + cg;
      union { uint4 q[2]; u16 u[16]; } bb;
      bb.q[0] = ((const uint4*)src)[0];
      bb.q[1] = ((const uint4*)src)[1];
      if (k0 + 32 < K) __builtin_prefetch(Bw + (size_t)(k0 + 32 + kk) * N + bn + cg, 0, 1);
#pragma unroll
      for (int c = 0; c < 16; c++) Bs[(cg + c) * 40 + kk] = bb.u[c];
    }
#if USE_ASYNC_LDS
    __builtin_amdgcn_s_wait_asynccnt(0);
#endif
    __syncthreads();

    AFrag af[4], bfr[2];
#pragma unroll
    for (int mt = 0; mt < 4; mt++) {
      int r = wm * 64 + mt * 16 + l16;
      af[mt].q[0] = *(const uint4*)&As[r * 40 + hf * 8];
      af[mt].q[1] = *(const uint4*)&As[r * 40 + 16 + hf * 8];
    }
#pragma unroll
    for (int nt = 0; nt < 2; nt++) {
      int c = wn * 32 + nt * 16 + l16;
      bfr[nt].q[0] = *(const uint4*)&Bs[c * 40 + hf * 8];
      bfr[nt].q[1] = *(const uint4*)&Bs[c * 40 + 16 + hf * 8];
    }
#pragma unroll
    for (int mt = 0; mt < 4; mt++)
#pragma unroll
      for (int nt = 0; nt < 2; nt++)
        acc[mt][nt] = wmma_bf16(af[mt].v, bfr[nt].v, acc[mt][nt]);
    __syncthreads();
  }

  // epilogue
#pragma unroll
  for (int mt = 0; mt < 4; mt++) {
#pragma unroll
    for (int nt = 0; nt < 2; nt++) {
      int col = bn + wn * 32 + nt * 16 + l16;
      float bv = (mode == 0) ? 0.f : bias[col];
#pragma unroll
      for (int v = 0; v < 8; v++) {
        int row = bm + wm * 64 + mt * 16 + hf * 8 + v;
        if (row < M) {
          float val = acc[mt][nt][v] + bv;
          if (mode == 2) val = 0.5f * val * (1.0f + erff(val * 0.70710678118654752f));
          size_t idx = (size_t)row * N + col;
          if (mode == 5) val += res[idx];
          if (mode >= 3) ((float*)Cout)[idx] = val;
          else           ((u16*)Cout)[idx] = f2bf(val);
        }
      }
    }
  }
}

// ---------------------------------------------------------------- temporal attention
// One wave per (seq in 1568, head in 12). Sequence length T=16, head_dim 64.
// qkv row layout: [q(12x64) | k(12x64) | v(12x64)] = 2304 cols, row = seq*16 + t.
__global__ __launch_bounds__(32)
void k_attn_t(const u16* __restrict__ qkv, u16* __restrict__ outp) {
  int blk = blockIdx.x;
  int s = blk / 12, h = blk - s * 12;
  int lane = threadIdx.x, hf = lane >> 4, l16 = lane & 15;
  const size_t ld = 2304;
  const size_t rowbase = (size_t)s * 16;

  __shared__ u16 P[16 * 40];
  __shared__ u16 V[32 * 72];
  __shared__ float sums[16];

  // zero the K-padding region of P (cols 16..31)
  for (int i = lane; i < 16 * 8; i += 32) {
    int row = i >> 3, uu = i & 7;
    *(u32*)&P[row * 40 + 16 + uu * 2] = 0u;
  }

  AFrag qf[2], kf[2];
#pragma unroll
  for (int c = 0; c < 2; c++) {
    const u16* p = qkv + (rowbase + l16) * ld + h * 64 + c * 32;
    qf[c].q[0] = *(const uint4*)(p + hf * 8);
    qf[c].q[1] = *(const uint4*)(p + 16 + hf * 8);
    const u16* pk = qkv + (rowbase + l16) * ld + 768 + h * 64 + c * 32;
    kf[c].q[0] = *(const uint4*)(pk + hf * 8);
    kf[c].q[1] = *(const uint4*)(pk + 16 + hf * 8);
  }
  const f32x8 z8 = {0.f, 0.f, 0.f, 0.f, 0.f, 0.f, 0.f, 0.f};
  f32x8 sc = z8;
  sc = wmma_bf16(qf[0].v, kf[0].v, sc);
  sc = wmma_bf16(qf[1].v, kf[1].v, sc);

  // softmax over the 16 key columns (per row); lane holds col=l16, rows hf*8+v
#pragma unroll
  for (int v = 0; v < 8; v++) {
    float val = sc[v] * 0.125f;
    float mx = rmax16(val);
    float e = expf(val - mx);
    float rs = rsum16(e);
    P[(hf * 8 + v) * 40 + l16] = f2bf(e);
    if (l16 == 0) sums[hf * 8 + v] = rs;
  }

  // stage V (16 real tokens, pad to 32 with zeros)
  for (int i = lane; i < 32 * 8; i += 32) {
    int t = i >> 3, seg = i & 7;
    uint4 d;
    d.x = d.y = d.z = d.w = 0u;
    if (t < 16) d = *(const uint4*)(qkv + (rowbase + t) * ld + 1536 + h * 64 + seg * 8);
    *(uint4*)&V[t * 72 + seg * 8] = d;
  }
  __syncthreads();

  AFrag pf;
  pf.q[0] = *(const uint4*)&P[l16 * 40 + hf * 8];
  pf.q[1] = *(const uint4*)&P[l16 * 40 + 16 + hf * 8];

  f32x8 o[4];
#pragma unroll
  for (int nt = 0; nt < 4; nt++) {
    AFrag vf;
    int n = nt * 16 + l16;
#pragma unroll
    for (int j = 0; j < 8; j++) {
      vf.us[j]     = V[(hf * 8 + j) * 72 + n];
      vf.us[8 + j] = V[(16 + hf * 8 + j) * 72 + n];
    }
    o[nt] = z8;
    o[nt] = wmma_bf16(pf.v, vf.v, o[nt]);
  }
#pragma unroll
  for (int nt = 0; nt < 4; nt++)
#pragma unroll
    for (int v = 0; v < 8; v++) {
      int row = hf * 8 + v;
      float val = o[nt][v] / sums[row];
      outp[(rowbase + row) * 768 + h * 64 + nt * 16 + l16] = f2bf(val);
    }
}

// ---------------------------------------------------------------- spatial attention (flash)
// One wave per (seq in 128, head in 12, qtile in 13). Seq len 197 (pad 224), head_dim 64.
__global__ __launch_bounds__(32)
void k_attn_s(const u16* __restrict__ qkv, u16* __restrict__ outp) {
  int blk = blockIdx.x;
  int qt = blk % 13;
  int sh = blk / 13;
  int s = sh / 12, h = sh - s * 12;
  int lane = threadIdx.x, hf = lane >> 4, l16 = lane & 15;
  const size_t ld = 2304;
  const size_t rowbase = (size_t)s * 197;

  int qrow = qt * 16 + l16;
  int qr = qrow < 197 ? qrow : 196;
  AFrag qf[2];
#pragma unroll
  for (int c = 0; c < 2; c++) {
    const u16* p = qkv + (rowbase + qr) * ld + h * 64 + c * 32;
    qf[c].q[0] = *(const uint4*)(p + hf * 8);
    qf[c].q[1] = *(const uint4*)(p + 16 + hf * 8);
  }

  __shared__ u16 P[16 * 40];
  __shared__ u16 V[32 * 72];

  const f32x8 z8 = {0.f, 0.f, 0.f, 0.f, 0.f, 0.f, 0.f, 0.f};
  f32x8 o[4];
  float m_i[8], l_i[8];
#pragma unroll
  for (int nt = 0; nt < 4; nt++) o[nt] = z8;
#pragma unroll
  for (int v = 0; v < 8; v++) { m_i[v] = -3.0e38f; l_i[v] = 0.f; }

  for (int kc = 0; kc < 7; kc++) {
    int t0 = kc * 32;
    f32x8 st[2];
#pragma unroll
    for (int kt = 0; kt < 2; kt++) {
      int kcol = t0 + kt * 16 + l16;
      int kr = kcol < 197 ? kcol : 0;
      AFrag kf0, kf1;
      const u16* p0 = qkv + (rowbase + kr) * ld + 768 + h * 64;
      kf0.q[0] = *(const uint4*)(p0 + hf * 8);
      kf0.q[1] = *(const uint4*)(p0 + 16 + hf * 8);
      const u16* p1 = p0 + 32;
      kf1.q[0] = *(const uint4*)(p1 + hf * 8);
      kf1.q[1] = *(const uint4*)(p1 + 16 + hf * 8);
      f32x8 sx = z8;
      sx = wmma_bf16(qf[0].v, kf0.v, sx);
      sx = wmma_bf16(qf[1].v, kf1.v, sx);
      st[kt] = sx;
    }
    bool ok0 = (t0 + l16) < 197;
    bool ok1 = (t0 + 16 + l16) < 197;
    float alpha[8];
#pragma unroll
    for (int v = 0; v < 8; v++) {
      float a0 = ok0 ? st[0][v] * 0.125f : -3.0e38f;
      float a1 = ok1 ? st[1][v] * 0.125f : -3.0e38f;
      float cm = rmax16(fmaxf(a0, a1));
      float mn = fmaxf(m_i[v], cm);
      float al = expf(m_i[v] - mn);
      float e0 = ok0 ? expf(a0 - mn) : 0.f;
      float e1 = ok1 ? expf(a1 - mn) : 0.f;
      float rs = rsum16(e0 + e1);
      m_i[v] = mn;
      l_i[v] = l_i[v] * al + rs;
      alpha[v] = al;
      P[(hf * 8 + v) * 40 + l16]      = f2bf(e0);
      P[(hf * 8 + v) * 40 + 16 + l16] = f2bf(e1);
    }
#pragma unroll
    for (int nt = 0; nt < 4; nt++)
#pragma unroll
      for (int v = 0; v < 8; v++) o[nt][v] *= alpha[v];

    // stage V chunk (tokens t0..t0+31, zero-pad beyond 197)
    for (int i = lane; i < 32 * 8; i += 32) {
      int t = i >> 3, seg = i & 7;
      int tok = t0 + t;
      uint4 d;
      d.x = d.y = d.z = d.w = 0u;
      if (tok < 197) d = *(const uint4*)(qkv + (rowbase + tok) * ld + 1536 + h * 64 + seg * 8);
      *(uint4*)&V[t * 72 + seg * 8] = d;
    }
    __syncthreads();

    AFrag pf;
    pf.q[0] = *(const uint4*)&P[l16 * 40 + hf * 8];
    pf.q[1] = *(const uint4*)&P[l16 * 40 + 16 + hf * 8];
#pragma unroll
    for (int nt = 0; nt < 4; nt++) {
      AFrag vf;
      int n = nt * 16 + l16;
#pragma unroll
      for (int j = 0; j < 8; j++) {
        vf.us[j]     = V[(hf * 8 + j) * 72 + n];
        vf.us[8 + j] = V[(16 + hf * 8 + j) * 72 + n];
      }
      o[nt] = wmma_bf16(pf.v, vf.v, o[nt]);
    }
    __syncthreads();
  }

#pragma unroll
  for (int nt = 0; nt < 4; nt++)
#pragma unroll
    for (int v = 0; v < 8; v++) {
      int row = qt * 16 + hf * 8 + v;
      if (row < 197) {
        float val = o[nt][v] / l_i[v];
        outp[(rowbase + row) * 768 + h * 64 + nt * 16 + l16] = f2bf(val);
      }
    }
}

// ---------------------------------------------------------------- elementwise
// xt2[b,n] = x[b,1+n] + tfc_out[b,n]
__global__ __launch_bounds__(256)
void k_add_xt2(const float* __restrict__ x, const float* __restrict__ t,
               float* __restrict__ xt2) {
  size_t i = (size_t)blockIdx.x * 256 + threadIdx.x;
  const size_t total = (size_t)25088 * 768;
  if (i >= total) return;
  int r = (int)(i / 768), c = (int)(i - (size_t)r * 768);
  int b = r / 3136, n = r - b * 3136;
  xt2[i] = x[((size_t)b * 3137 + 1 + n) * 768 + c] + t[i];
}

// x_new assembly: one block per (b,tok) row of (8,3137)
__global__ __launch_bounds__(256)
void k_assemble(const float* __restrict__ x, const float* __restrict__ xt2,
                const float* __restrict__ rs, float* __restrict__ xnew) {
  int r = blockIdx.x;
  int b = r / 3137, tok = r - b * 3137;
  int tid = threadIdx.x;
#pragma unroll
  for (int e = 0; e < 3; e++) {
    int c = tid + e * 256;
    float v;
    if (tok == 0) {
      float sum = 0.f;
#pragma unroll
      for (int t = 0; t < 16; t++)
        sum += rs[(((size_t)b * 16 + t) * 197) * 768 + c];
      v = x[(size_t)b * 3137 * 768 + c] + sum * (1.0f / 16.0f);
    } else {
      int n = tok - 1, hw = n >> 4, t = n & 15;
      v = xt2[((size_t)b * 3136 + n) * 768 + c] +
          rs[(((size_t)b * 16 + t) * 197 + 1 + hw) * 768 + c];
    }
    xnew[(size_t)r * 768 + c] = v;
  }
}

// ---------------------------------------------------------------- launch
extern "C" void kernel_launch(void* const* d_in, const int* in_sizes, int n_in,
                              void* d_out, int out_size, void* d_ws, size_t ws_size,
                              hipStream_t stream) {
  (void)in_sizes; (void)n_in; (void)out_size; (void)ws_size;
  const float* x        = (const float*)d_in[0];
  const float* tn1_w    = (const float*)d_in[1];
  const float* tn1_b    = (const float*)d_in[2];
  const float* t_qkv_w  = (const float*)d_in[3];
  const float* t_proj_w = (const float*)d_in[4];
  const float* t_proj_b = (const float*)d_in[5];
  const float* tfc_w    = (const float*)d_in[6];
  const float* tfc_b    = (const float*)d_in[7];
  const float* n1_w     = (const float*)d_in[8];
  const float* n1_b     = (const float*)d_in[9];
  const float* s_qkv_w  = (const float*)d_in[10];
  const float* s_proj_w = (const float*)d_in[11];
  const float* s_proj_b = (const float*)d_in[12];
  const float* n2_w     = (const float*)d_in[13];
  const float* n2_b     = (const float*)d_in[14];
  const float* fc1_w    = (const float*)d_in[15];
  const float* fc1_b    = (const float*)d_in[16];
  const float* fc2_w    = (const float*)d_in[17];
  const float* fc2_b    = (const float*)d_in[18];
  float* outF = (float*)d_out;

  const int MT = 25088;   // 8*3136  temporal rows
  const int MS = 25216;   // 128*197 spatial rows
  const int MF = 25096;   // 8*3137  full rows

  // workspace carve-up (~510 MB total)
  char* ws = (char*)d_ws;
  size_t off = 0;
  auto alloc = [&](size_t bytes) -> void* {
    void* p = (void*)(ws + off);
    off += (bytes + 255) & ~(size_t)255;
    return p;
  };
  u16* wqt  = (u16*)alloc((size_t)768 * 2304 * 2);
  u16* wpt  = (u16*)alloc((size_t)768 * 768 * 2);
  u16* wfct = (u16*)alloc((size_t)768 * 768 * 2);
  u16* wqs  = (u16*)alloc((size_t)768 * 2304 * 2);
  u16* wps  = (u16*)alloc((size_t)768 * 768 * 2);
  u16* wfc1 = (u16*)alloc((size_t)768 * 3072 * 2);
  u16* wfc2 = (u16*)alloc((size_t)3072 * 768 * 2);
  u16* lnb  = (u16*)alloc((size_t)MS * 768 * 2);
  u16* qkvb = (u16*)alloc((size_t)MS * 2304 * 2);
  u16* attb = (u16*)alloc((size_t)MS * 768 * 2);
  u16* prjb = (u16*)alloc((size_t)MT * 768 * 2);
  float* xt2  = (float*)alloc((size_t)MT * 768 * 4);
  float* xnew = (float*)alloc((size_t)MF * 768 * 4);
  float* tmpf = (float*)alloc((size_t)MS * 768 * 4);
  u16* hid  = (u16*)alloc((size_t)4096 * 3072 * 2);

  auto cvt = [&](const float* s, u16* d, int n) {
    k_cvt<<<dim3((n + 255) / 256), dim3(256), 0, stream>>>(s, d, n);
  };
  auto gemm = [&](const u16* A, const u16* B, const float* bias, const float* res,
                  void* C, int M, int N, int K, int mode) {
    dim3 g((M + 127) / 128, N / 128);
    k_gemm<<<g, dim3(256), 0, stream>>>(A, B, bias, res, C, M, N, K, mode);
  };

  // weights -> bf16
  cvt(t_qkv_w,  wqt,  768 * 2304);
  cvt(t_proj_w, wpt,  768 * 768);
  cvt(tfc_w,    wfct, 768 * 768);
  cvt(s_qkv_w,  wqs,  768 * 2304);
  cvt(s_proj_w, wps,  768 * 768);
  cvt(fc1_w,    wfc1, 768 * 3072);
  cvt(fc2_w,    wfc2, 3072 * 768);

  // ---- temporal branch
  k_ln<<<dim3(MT), dim3(256), 0, stream>>>(x, nullptr, tn1_w, tn1_b, lnb, 1);
  gemm(lnb, wqt, nullptr, nullptr, qkvb, MT, 2304, 768, 0);
  k_attn_t<<<dim3(1568 * 12), dim3(32), 0, stream>>>(qkvb, attb);
  gemm(attb, wpt, t_proj_b, nullptr, prjb, MT, 768, 768, 1);
  gemm(prjb, wfct, tfc_b, nullptr, tmpf, MT, 768, 768, 3);
  k_add_xt2<<<dim3(((size_t)MT * 768 + 255) / 256), dim3(256), 0, stream>>>(x, tmpf, xt2);

  // ---- spatial branch
  k_ln<<<dim3(MS), dim3(256), 0, stream>>>(x, xt2, n1_w, n1_b, lnb, 2);
  gemm(lnb, wqs, nullptr, nullptr, qkvb, MS, 2304, 768, 0);
  k_attn_s<<<dim3(128 * 12 * 13), dim3(32), 0, stream>>>(qkvb, attb);
  gemm(attb, wps, s_proj_b, nullptr, tmpf, MS, 768, 768, 3);
  k_assemble<<<dim3(MF), dim3(256), 0, stream>>>(x, xt2, tmpf, xnew);

  // ---- MLP (chunked so the 3072-wide hidden fits a 25 MB slab)
  k_ln<<<dim3(MF), dim3(256), 0, stream>>>(xnew, nullptr, n2_w, n2_b, lnb, 0);
  const int CH = 4096;
  for (int m0 = 0; m0 < MF; m0 += CH) {
    int mc = (MF - m0) < CH ? (MF - m0) : CH;
    gemm(lnb + (size_t)m0 * 768, wfc1, fc1_b, nullptr, hid, mc, 3072, 768, 2);
    gemm(hid, wfc2, fc2_b, xnew + (size_t)m0 * 768, outF + (size_t)m0 * 768,
         mc, 768, 3072, 5);
  }
}
